// MyGRU_34462817583400
// MI455X (gfx1250) — matrix operand order repaired and checked by reading, not compiled, not run on hardware
//
#include <hip/hip_runtime.h>

// ---------------------------------------------------------------------------
// CDNA5 GRU: bf16 WMMA GEMMs, f32 state/accumulate.
//   K0: embed gather -> Xemb bf16 [32768, 320] (E padded 300->320), (s,b) rows
//   K1..K4: pack recurrent/x weights into WMMA-B lane layout + bias vector
//   K5: Xpre[t][b][n] = Xemb @ Wx + bias  (big parallel WMMA GEMM, bf16 out)
//   K6: persistent recurrent scan (32 blocks), grid-sync via atomics,
//       2 dependent WMMA GEMMs per step; constant B-matrices cached in LDS
//       (256 KB/WGP of the 320 KB available), next-step Xpre prefetched
//   K7: tiny dense [64,512]@[512,3]
// ---------------------------------------------------------------------------

typedef __attribute__((ext_vector_type(16))) __bf16 v16bf;
typedef __attribute__((ext_vector_type(8)))  float  v8f;

union BV { uint4 u4[2]; v16bf v; };

__device__ __forceinline__ unsigned short f2bf(float f) {
  unsigned u = __float_as_uint(f);
  unsigned r = (u + 0x7FFFu + ((u >> 16) & 1u)) >> 16;   // round-to-nearest-even
  return (unsigned short)r;
}
__device__ __forceinline__ float bf2f(unsigned short h) {
  return __uint_as_float(((unsigned)h) << 16);
}

// A-tile (16x32 bf16) from row-major src (already offset to m0,k0), stride lda halves.
// Layout: lanes 0-15 row r hold K 0..15; lanes 16-31 hold K 16..31 (matches packed B).
__device__ __forceinline__ v16bf load_a(const unsigned short* src, int lda, int lane) {
  const int r = lane & 15, half = lane >> 4;
  const unsigned short* p = src + (size_t)r * lda + 16 * half;
  BV x;
  x.u4[0] = *(const uint4*)(p);
  x.u4[1] = *(const uint4*)(p + 8);
  return x.v;
}

// B-tile (32x16 bf16) pre-packed: each lane's 16 halves are contiguous.
__device__ __forceinline__ v16bf load_b(const unsigned short* tile, int lane) {
  const unsigned short* p = tile + lane * 16;
  BV x;
  x.u4[0] = *(const uint4*)(p);
  x.u4[1] = *(const uint4*)(p + 8);
  return x.v;
}

#define WMMA_BF16(a, b, c) \
  __builtin_amdgcn_wmma_f32_16x16x32_bf16(false, (a), false, (b), (short)0, (c), false, false)

// --------------------------- constants -------------------------------------
#define B_   64
#define S_   512
#define H_   512
#define E_   300
#define EP_  320      // E padded to multiple of 32
#define N3_  1536     // 3*H
#define NWG_ 32       // persistent blocks in recurrent kernel

// LDS layout for the recurrent kernel (halves): 8 waves x (16 tiles x 512 halves)
// phase-A strips at [wv*8192], phase-B strips at [65536 + wv*8192]; 256 KB total.
#define LDS_HALVES 131072

// --------------------------- K0: embed gather ------------------------------
__global__ void k_embed(const int* __restrict__ text, const float* __restrict__ table,
                        unsigned short* __restrict__ xemb) {
  int idx = blockIdx.x * 256 + threadIdx.x;           // row*80 + chunk
  if (idx >= (B_ * S_) * (EP_ / 4)) return;
  int row = idx / (EP_ / 4), c = idx % (EP_ / 4);     // row = s*64 + b
  int s = row >> 6, b = row & 63;
  int k0 = c * 4;
  unsigned long long out = 0ull;
  if (k0 < E_) {                                       // 300 % 4 == 0 -> clean edge
    int tok = text[b * S_ + s];
    float4 f = *(const float4*)(table + (size_t)tok * E_ + k0);
    out = (unsigned long long)f2bf(f.x)
        | ((unsigned long long)f2bf(f.y) << 16)
        | ((unsigned long long)f2bf(f.z) << 32)
        | ((unsigned long long)f2bf(f.w) << 48);
  }
  *(unsigned long long*)(xemb + (size_t)row * EP_ + k0) = out;
}

// --------------------------- K1..K4: weight packing ------------------------
// dst element d -> tile t=d>>9, within-tile r=d&511, lane l=r>>4, half j=r&15
// K = kt*32 + 16*(l>>4) + j ; N = nt*16 + (l&15)
__global__ void k_pack_wzr(const float* __restrict__ Wz, const float* __restrict__ Wr,
                           unsigned short* __restrict__ dst) {
  int d = blockIdx.x * 256 + threadIdx.x;
  if (d >= H_ * 2 * H_) return;
  int t = d >> 9, r = d & 511, l = r >> 4, j = r & 15;
  int nt = t >> 4, kt = t & 15;
  int K = kt * 32 + 16 * (l >> 4) + j;
  int N = nt * 16 + (l & 15);
  float v = (N < H_) ? Wz[(size_t)K * H_ + N] : Wr[(size_t)K * H_ + (N - H_)];
  dst[d] = f2bf(v);
}

__global__ void k_pack_wh(const float* __restrict__ Wh, unsigned short* __restrict__ dst) {
  int d = blockIdx.x * 256 + threadIdx.x;
  if (d >= H_ * H_) return;
  int t = d >> 9, r = d & 511, l = r >> 4, j = r & 15;
  int nt = t >> 4, kt = t & 15;
  int K = kt * 32 + 16 * (l >> 4) + j;
  int N = nt * 16 + (l & 15);
  dst[d] = f2bf(Wh[(size_t)K * H_ + N]);
}

__global__ void k_pack_wx(const float* __restrict__ Wz, const float* __restrict__ Wr,
                          const float* __restrict__ Wh, unsigned short* __restrict__ dst) {
  int d = blockIdx.x * 256 + threadIdx.x;
  if (d >= EP_ * N3_) return;
  int t = d >> 9, r = d & 511, l = r >> 4, j = r & 15;
  int nt = t / 10, kt = t % 10;                        // 10 K-tiles (EP_/32)
  int K = kt * 32 + 16 * (l >> 4) + j;
  int N = nt * 16 + (l & 15);
  float v = 0.f;
  if (K < E_) {
    int g = N >> 9, col = N & 511;
    const float* W = (g == 0) ? Wz : ((g == 1) ? Wr : Wh);
    v = W[(size_t)(H_ + K) * H_ + col];                // rows H..H+E-1 are the x part
  }
  dst[d] = f2bf(v);
}

__global__ void k_pack_bias(const float* __restrict__ bz, const float* __restrict__ br,
                            const float* __restrict__ bh, float* __restrict__ dst) {
  int n = blockIdx.x * 256 + threadIdx.x;
  if (n >= N3_) return;
  dst[n] = (n < H_) ? bz[n] : ((n < 2 * H_) ? br[n - H_] : bh[n - 2 * H_]);
}

// --------------------------- K5: X projections -----------------------------
// [32768, 320] @ [320, 1536] + bias -> Xpre bf16 [t][b][n] (row = s*64+b)
__global__ __launch_bounds__(256) void k_xproj(const unsigned short* __restrict__ xemb,
                                               const unsigned short* __restrict__ wx,
                                               const float* __restrict__ bias,
                                               unsigned short* __restrict__ xpre) {
  const int lane = threadIdx.x & 31, wv = threadIdx.x >> 5;
  const int mt = blockIdx.y * 8 + wv;                  // 0..2047
  const int nt = blockIdx.x;                           // 0..95
  const int m0 = mt * 16;
  const int col = nt * 16 + (lane & 15);
  v8f c;
  const float bv = bias[col];
#pragma unroll
  for (int j = 0; j < 8; ++j) c[j] = bv;
#pragma unroll
  for (int kt = 0; kt < EP_ / 32; ++kt) {
    v16bf a = load_a(xemb + (size_t)m0 * EP_ + kt * 32, EP_, lane);
    v16bf bm = load_b(wx + (((size_t)nt * 10 + kt) << 9), lane);
    c = WMMA_BF16(a, bm, c);
  }
#pragma unroll
  for (int j = 0; j < 8; ++j) {
    int row = m0 + j + 8 * (lane >> 4);                // = s*64 + b
    xpre[(size_t)row * N3_ + col] = f2bf(c[j]);
  }
}

// --------------------------- grid-wide barrier -----------------------------
__device__ __forceinline__ void gridsync(unsigned* cnt, unsigned target) {
  __syncthreads();
  if (threadIdx.x == 0) {
    __threadfence();                                   // release prior stores
    atomicAdd(cnt, 1u);
    volatile unsigned* vc = (volatile unsigned*)cnt;
    while (*vc < target) { __builtin_amdgcn_s_sleep(1); }
  }
  __syncthreads();
  __threadfence();                                     // acquire others' stores
}

// --------------------------- K6: recurrent scan ----------------------------
__global__ __launch_bounds__(256) void k_recurrent(
    const unsigned short* __restrict__ wzr, const unsigned short* __restrict__ wh,
    const unsigned short* __restrict__ xpre,
    float* h_f32, unsigned short* h_bf16, float* z_f32, unsigned short* rh_bf16,
    float* ylast, unsigned* sync) {
  extern __shared__ __align__(16) unsigned short smem[];   // 256 KB dynamic LDS
  const int lane = threadIdx.x & 31, wv = threadIdx.x >> 5;
  const int w = blockIdx.x * 8 + wv;                   // 0..255 global wave id

  const int mt = w & 3, ntA = w >> 2;                  // phase A: 4 x 64 tiles
  const int m0 = mt * 16;
  const int colA = ntA * 16 + (lane & 15);             // 0..1023 (z|r)
  const int half = lane >> 4;
  const int nt2  = w >> 2;                             // phase B tile col (w<128)
  const int colB = nt2 * 16 + (lane & 15);

  // ---- prefill LDS with this wave's constant B strips (once per launch) ----
  const int baseA = wv * 8192;                         // 16 tiles * 512 halves
  const int baseB = 65536 + wv * 8192;
#pragma unroll
  for (int kt = 0; kt < 16; ++kt) {
    const unsigned short* src = wzr + (((size_t)ntA * 16 + kt) << 9) + lane * 16;
    unsigned short* dstl = &smem[baseA + kt * 512 + lane * 16];
    *(uint4*)(dstl)     = *(const uint4*)(src);
    *(uint4*)(dstl + 8) = *(const uint4*)(src + 8);
  }
  if (blockIdx.x < 16) {                               // blocks owning phase-B tiles
#pragma unroll
    for (int kt = 0; kt < 16; ++kt) {
      const unsigned short* src = wh + (((size_t)nt2 * 16 + kt) << 9) + lane * 16;
      unsigned short* dstl = &smem[baseB + kt * 512 + lane * 16];
      *(uint4*)(dstl)     = *(const uint4*)(src);
      *(uint4*)(dstl + 8) = *(const uint4*)(src + 8);
    }
  }
  __syncthreads();

  // zero-init hidden state
  for (int i = blockIdx.x * 256 + threadIdx.x; i < B_ * H_; i += NWG_ * 256) {
    h_f32[i] = 0.f; h_bf16[i] = 0;
  }
  unsigned bar = 0;
  gridsync(sync, ++bar * NWG_);

  for (int t = 0; t < S_; ++t) {
    const unsigned short* xp  = xpre + (size_t)t * B_ * N3_;
    const unsigned short* xpn = xpre + (size_t)(t + 1 < S_ ? t + 1 : t) * B_ * N3_;

    // ---- phase A: [z|r]_pre = h @ Whh_zr + Xpre_zr ; sigmoid ; build r*h ----
    {
      v8f c;
#pragma unroll
      for (int j = 0; j < 8; ++j) {
        int rb = m0 + j + 8 * half;
        c[j] = bf2f(xp[(size_t)rb * N3_ + colA]);
        __builtin_prefetch(xpn + (size_t)rb * N3_ + colA, 0, 1);  // next step
      }
#pragma unroll
      for (int kt = 0; kt < 16; ++kt) {
        v16bf a = load_a(h_bf16 + (size_t)m0 * H_ + kt * 32, H_, lane);
        BV bx;
        const unsigned short* p = &smem[baseA + kt * 512 + lane * 16];
        bx.u4[0] = *(const uint4*)(p);
        bx.u4[1] = *(const uint4*)(p + 8);
        c = WMMA_BF16(a, bx.v, c);
      }
#pragma unroll
      for (int j = 0; j < 8; ++j) {
        int rb = m0 + j + 8 * half;
        float g = 1.f / (1.f + __expf(-c[j]));
        if (ntA < 32) {                                // z gate
          z_f32[rb * H_ + colA] = g;
        } else {                                       // r gate -> r*h (bf16)
          int hc = colA - H_;
          float hv = h_f32[rb * H_ + hc];
          rh_bf16[rb * H_ + hc] = f2bf(g * hv);
        }
      }
    }
    gridsync(sync, ++bar * NWG_);

    // ---- phase B: y = tanh((r*h) @ Whh_h + Xpre_h) ; h = (1-z)h + z y ------
    if (w < 128) {
      v8f c;
#pragma unroll
      for (int j = 0; j < 8; ++j) {
        int rb = m0 + j + 8 * half;
        c[j] = bf2f(xp[(size_t)rb * N3_ + 2 * H_ + colB]);
      }
#pragma unroll
      for (int kt = 0; kt < 16; ++kt) {
        v16bf a = load_a(rh_bf16 + (size_t)m0 * H_ + kt * 32, H_, lane);
        BV bx;
        const unsigned short* p = &smem[baseB + kt * 512 + lane * 16];
        bx.u4[0] = *(const uint4*)(p);
        bx.u4[1] = *(const uint4*)(p + 8);
        c = WMMA_BF16(a, bx.v, c);
      }
#pragma unroll
      for (int j = 0; j < 8; ++j) {
        int rb = m0 + j + 8 * half;
        float y  = tanhf(c[j]);
        float z  = z_f32[rb * H_ + colB];
        float hv = h_f32[rb * H_ + colB];
        float hn = (1.f - z) * hv + z * y;
        h_f32[rb * H_ + colB]  = hn;
        h_bf16[rb * H_ + colB] = f2bf(hn);
        if (t == S_ - 1) ylast[rb * H_ + colB] = y;
      }
    }
    gridsync(sync, ++bar * NWG_);
  }
}

// --------------------------- K7: final dense -------------------------------
__global__ void k_dense(const float* __restrict__ ylast, const float* __restrict__ dW,
                        const float* __restrict__ db, float* __restrict__ out) {
  int tid = blockIdx.x * blockDim.x + threadIdx.x;
  if (tid >= B_ * 3) return;
  int b = tid / 3, p = tid % 3;
  float s = db[p];
  for (int k = 0; k < H_; ++k) s += ylast[b * H_ + k] * dW[k * 3 + p];
  out[tid] = s;
}

// --------------------------- host launcher ---------------------------------
extern "C" void kernel_launch(void* const* d_in, const int* in_sizes, int n_in,
                              void* d_out, int out_size, void* d_ws, size_t ws_size,
                              hipStream_t stream) {
  const int*   text  = (const int*)  d_in[0];
  const float* table = (const float*)d_in[1];
  const float* Wz    = (const float*)d_in[2];
  const float* bz    = (const float*)d_in[3];
  const float* Wr    = (const float*)d_in[4];
  const float* br    = (const float*)d_in[5];
  const float* Wh    = (const float*)d_in[6];
  const float* bh    = (const float*)d_in[7];
  const float* dW    = (const float*)d_in[8];
  const float* db    = (const float*)d_in[9];
  float* out = (float*)d_out;

  char* ws = (char*)d_ws;
  size_t o = 0;
  auto alloc = [&](size_t bytes) { size_t r = o; o = (o + bytes + 255) & ~(size_t)255; return r; };
  size_t oSync  = alloc(128);
  size_t oHf    = alloc((size_t)B_ * H_ * 4);
  size_t oHb    = alloc((size_t)B_ * H_ * 2);
  size_t oZ     = alloc((size_t)B_ * H_ * 4);
  size_t oRH    = alloc((size_t)B_ * H_ * 2);
  size_t oY     = alloc((size_t)B_ * H_ * 4);
  size_t oWzr   = alloc((size_t)H_ * 2 * H_ * 2);
  size_t oWh    = alloc((size_t)H_ * H_ * 2);
  size_t oWx    = alloc((size_t)EP_ * N3_ * 2);
  size_t oBias  = alloc((size_t)N3_ * 4);
  size_t oXemb  = alloc((size_t)B_ * S_ * EP_ * 2);
  size_t oXpre  = alloc((size_t)S_ * B_ * N3_ * 2);
  if (ws_size < o) return;  // workspace too small: deterministic no-op

  unsigned*       sync  = (unsigned*)      (ws + oSync);
  float*          h_f32 = (float*)         (ws + oHf);
  unsigned short* h_b16 = (unsigned short*)(ws + oHb);
  float*          z_f32 = (float*)         (ws + oZ);
  unsigned short* rh_b  = (unsigned short*)(ws + oRH);
  float*          ylast = (float*)         (ws + oY);
  unsigned short* wzr   = (unsigned short*)(ws + oWzr);
  unsigned short* whp   = (unsigned short*)(ws + oWh);
  unsigned short* wxp   = (unsigned short*)(ws + oWx);
  float*          biasp = (float*)         (ws + oBias);
  unsigned short* xemb  = (unsigned short*)(ws + oXemb);
  unsigned short* xpre  = (unsigned short*)(ws + oXpre);

  hipMemsetAsync(sync, 0, 128, stream);

  k_embed<<<(B_ * S_ * (EP_ / 4) + 255) / 256, 256, 0, stream>>>(text, table, xemb);
  k_pack_wzr<<<(H_ * 2 * H_ + 255) / 256, 256, 0, stream>>>(Wz, Wr, wzr);
  k_pack_wh <<<(H_ * H_ + 255) / 256, 256, 0, stream>>>(Wh, whp);
  k_pack_wx <<<(EP_ * N3_ + 255) / 256, 256, 0, stream>>>(Wz, Wr, Wh, wxp);
  k_pack_bias<<<(N3_ + 255) / 256, 256, 0, stream>>>(bz, br, bh, biasp);

  dim3 gp(96, (B_ * S_) / 16 / 8);                     // (N tiles, M tile groups)
  k_xproj<<<gp, 256, 0, stream>>>(xemb, wxp, biasp, xpre);

  k_recurrent<<<NWG_, 256, LDS_HALVES * sizeof(unsigned short), stream>>>(
      wzr, whp, xpre, h_f32, h_b16, z_f32, rh_b, ylast, sync);

  k_dense<<<1, 256, 0, stream>>>(ylast, dW, db, out);
}
